// BiLstmCrf_33947421508246
// MI455X (gfx1250) — compile-verified, matrix operand order checked by reading, not run on hardware
//
#include <hip/hip_runtime.h>
#include <hip/hip_bf16.h>

// ---------------- problem constants ----------------
#define BB   256
#define TT   256
#define DC_  256
#define DF_  128
#define DD   384          // DC + DF
#define HH   512
#define FOURH 2048
#define KK   896          // DD + HH
#define NTAG 66
#define TSTART 64
#define TSTOP  65

typedef __attribute__((ext_vector_type(16))) __bf16 v16bf;
typedef __attribute__((ext_vector_type(8)))  float  v8f;

struct Pack32 { uint4 lo; uint4 hi; };   // 32 bytes == v16bf

static __device__ __forceinline__ unsigned short f2bf(float f) {
    unsigned u = __builtin_bit_cast(unsigned, f);
    unsigned r = u + 0x7FFFu + ((u >> 16) & 1u);   // round-to-nearest-even
    return (unsigned short)(r >> 16);
}
static __device__ __forceinline__ float sigm(float x) {
    return 1.0f / (1.0f + __expf(-x));
}
static __device__ __forceinline__ v16bf load_frag(const unsigned short* p) {
    Pack32 pk;
    pk.lo = *(const uint4*)(p);        // K = base .. base+7
    pk.hi = *(const uint4*)(p + 16);   // K = base+16 .. base+23
    return __builtin_bit_cast(v16bf, pk);
}
// CDNA5 async copy: GLOBAL -> LDS, 16 bytes per lane, tracked by ASYNCcnt
static __device__ __forceinline__ void async_g2l_b128(unsigned lds_off,
                                                      unsigned long long gaddr) {
    asm volatile("global_load_async_to_lds_b128 %0, %1, off"
                 :: "v"(lds_off), "v"(gaddr) : "memory");
}
static __device__ __forceinline__ void wait_asynccnt0() {
    asm volatile("s_wait_asynccnt 0x0" ::: "memory");
}

// ---------------- weight / embedding prep ----------------
__global__ void f32_to_bf16_k(const float* __restrict__ src,
                              unsigned short* __restrict__ dst, int n) {
    int i = blockIdx.x * blockDim.x + threadIdx.x;
    if (i < n) dst[i] = f2bf(src[i]);
}

// W_out (66 x 1024) -> padded (80 x 1024) bf16, zero rows 66..79
__global__ void conv_wout_k(const float* __restrict__ src,
                            unsigned short* __restrict__ dst) {
    int i = blockIdx.x * blockDim.x + threadIdx.x;
    if (i >= 80 * 1024) return;
    int row = i >> 10;
    dst[i] = (row < NTAG) ? f2bf(src[i]) : (unsigned short)0;
}

// x[b][t][k] = k<256 ? char_emb[input[b,t]][k] : feat_emb[feature[b]][k-256]
__global__ void embed_k(const int* __restrict__ inp, const int* __restrict__ feat,
                        const float* __restrict__ cemb, const float* __restrict__ femb,
                        unsigned short* __restrict__ xbf) {
    long long i = (long long)blockIdx.x * blockDim.x + threadIdx.x;
    if (i >= (long long)BB * TT * DD) return;
    int k  = (int)(i % DD);
    int bt = (int)(i / DD);
    int b  = bt >> 8;
    float v;
    if (k < DC_) v = cemb[(size_t)inp[bt] * DC_ + k];
    else         v = femb[(size_t)feat[b] * DF_ + (k - DC_)];
    xbf[i] = f2bf(v);
}

__global__ void init_state_k(float* __restrict__ hA, float* __restrict__ hB,
                             float* __restrict__ cA, float* __restrict__ cB,
                             unsigned short* __restrict__ hbfA,
                             unsigned short* __restrict__ hbfB,
                             float* __restrict__ loss) {
    int i = blockIdx.x * blockDim.x + threadIdx.x;   // 2*B*H = 262144
    if (i == 0) loss[0] = 0.0f;
    if (i < 2 * BB * HH) {
        hA[i] = 0.0f; hB[i] = 0.0f; cA[i] = 0.0f; cB[i] = 0.0f;
        hbfA[i] = 0; hbfB[i] = 0;
    }
}

// ---------------- fused LSTM step: GEMM + gates (one timestep, both dirs) ---
// Each block: 16 batch rows x 32 hidden columns, all 4 gate chunks.
// 8 waves = 4 gates x 2 column tiles of 16.  grid (16, 16, 2), block 256.
__global__ void lstm_step_fused_k(const unsigned short* __restrict__ xbf,
                                  const unsigned short* __restrict__ hbf_in,
                                  const unsigned short* __restrict__ wih_f,
                                  const unsigned short* __restrict__ whh_f,
                                  const unsigned short* __restrict__ wih_b,
                                  const unsigned short* __restrict__ whh_b,
                                  const float* __restrict__ bias_f,
                                  const float* __restrict__ bias_b,
                                  const int* __restrict__ blen,
                                  const float* __restrict__ h_in,
                                  const float* __restrict__ c_in,
                                  float* __restrict__ h_out,
                                  float* __restrict__ c_out,
                                  unsigned short* __restrict__ hbf_out,
                                  unsigned short* __restrict__ out_f,
                                  unsigned short* __restrict__ out_b,
                                  int t) {
    __shared__ alignas(16) unsigned short As[16 * KK];   // 28 KB A panel (bf16)
    __shared__ float sG[4 * 16 * 32];                    // 8 KB gate staging
    const int tid = threadIdx.x;
    const int dir = blockIdx.z;
    const int m0  = blockIdx.y * 16;

    // --- async fill of A panel: rows = 16 batch entries, K = 896 (x_t | h) ---
    for (int q = tid; q < 16 * (KK / 8); q += 256) {
        int row = q / (KK / 8);
        int k   = (q - row * (KK / 8)) * 8;
        int b   = m0 + row;
        const unsigned short* src;
        if (k < DD) {
            int tx = t;
            if (dir) { int r = blen[b] - 1 - t; tx = (r >= 0) ? r : t; }
            src = xbf + ((size_t)(b * TT + tx)) * DD + k;
        } else {
            src = hbf_in + ((size_t)(dir * BB + b)) * HH + (k - DD);
        }
        unsigned lds = (unsigned)(unsigned long long)(const void*)(As + row * KK + k);
        async_g2l_b128(lds, (unsigned long long)src);
    }
    wait_asynccnt0();
    __syncthreads();

    const int wave  = tid >> 5;
    const int lane  = tid & 31;
    const int gate  = wave >> 1;                     // 0..3 (i,f,g,o)
    const int ct    = wave & 1;                      // column sub-tile
    const int ncol  = blockIdx.x * 32 + ct * 16;     // within gate chunk [0,512)
    const int nb    = gate * HH + ncol + (lane & 15);// row of [4H x K] weight
    const int half8 = (lane >> 4) * 8;
    const int arow  = lane & 15;
    const unsigned short* wih = dir ? wih_b : wih_f;
    const unsigned short* whh = dir ? whh_b : whh_f;

    v8f acc = {};
    #pragma unroll 4
    for (int k0 = 0; k0 < KK; k0 += 32) {
        v16bf a = load_frag(As + arow * KK + k0 + half8);
        const unsigned short* bp =
            (k0 < DD) ? (wih + (size_t)nb * DD + k0 + half8)
                      : (whh + (size_t)nb * HH + (k0 - DD) + half8);
        v16bf bt = load_frag(bp);
        acc = __builtin_amdgcn_wmma_f32_16x16x32_bf16(
            false, a, false, bt, (short)0, acc, false, false);
    }

    // stage gate pre-activations: sG[gate][row 0..15][col 0..31]
    {
        const int col = ct * 16 + (lane & 15);
        #pragma unroll
        for (int r = 0; r < 8; ++r) {
            int row = r + (lane >> 4) * 8;
            sG[(gate * 16 + row) * 32 + col] = acc[r];
        }
    }
    __syncthreads();

    // --- pointwise gates: 16 rows x 32 cols = 512 elements ---
    const float* bias = dir ? bias_b : bias_f;
    for (int e = tid; e < 512; e += 256) {
        int row = e >> 5, col = e & 31;
        int j   = blockIdx.x * 32 + col;             // hidden index [0,512)
        int b   = m0 + row;
        int s   = row * 32 + col;
        float gi = sG[0 * 512 + s] + bias[j];
        float gf = sG[1 * 512 + s] + bias[HH + j];
        float gg = sG[2 * 512 + s] + bias[2 * HH + j];
        float go = sG[3 * 512 + s] + bias[3 * HH + j];
        size_t hidx = ((size_t)(dir * BB + b)) * HH + j;
        float cp = c_in[hidx], hp = h_in[hidx];
        float cn = sigm(gf) * cp + sigm(gi) * tanhf(gg);
        float hn = sigm(go) * tanhf(cn);
        bool  mt = (t < blen[b]);
        float hw = mt ? hn : hp;
        h_out[hidx] = hw;
        c_out[hidx] = mt ? cn : cp;
        hbf_out[hidx] = f2bf(hw);
        unsigned short ob = f2bf(mt ? hn : 0.0f);
        if (dir == 0) out_f[((size_t)b * TT + t) * HH + j] = ob;
        else          out_b[((size_t)b * TT + t) * HH + j] = ob;  // reversed time
    }
}

// ---------------- emission projection: feats = [out_f|out_b] @ W_out.T + b --
// grid: (512 blocks of 8 M-tiles, 5 N-tiles), block = 256 (8 waves)
__global__ void emissions_gemm_k(const unsigned short* __restrict__ out_f,
                                 const unsigned short* __restrict__ out_b,
                                 const unsigned short* __restrict__ woutb, // [80][1024]
                                 const float* __restrict__ bout,
                                 const int* __restrict__ blen,
                                 float* __restrict__ feats) {
    const int tid   = threadIdx.x;
    const int wave  = tid >> 5, lane = tid & 31;
    const int mtile = blockIdx.x * 8 + wave;       // 0..4095
    const int rowbase = mtile * 16;
    const int n0    = blockIdx.y * 16;
    const int half8 = (lane >> 4) * 8;
    const int arow  = rowbase + (lane & 15);
    const int b = arow >> 8, t = arow & 255;
    int tb = t; { int r = blen[b] - 1 - t; if (r >= 0) tb = r; }
    const unsigned short* af = out_f + ((size_t)b * TT + t)  * HH;
    const unsigned short* ab = out_b + ((size_t)b * TT + tb) * HH;
    const unsigned short* wrow = woutb + (size_t)(n0 + (lane & 15)) * 1024;

    v8f acc = {};
    #pragma unroll 4
    for (int k0 = 0; k0 < 1024; k0 += 32) {
        const unsigned short* ap =
            (k0 < HH) ? (af + k0 + half8) : (ab + (k0 - HH) + half8);
        v16bf a  = load_frag(ap);
        v16bf bt = load_frag(wrow + k0 + half8);
        acc = __builtin_amdgcn_wmma_f32_16x16x32_bf16(
            false, a, false, bt, (short)0, acc, false, false);
    }
    const int n = n0 + (lane & 15);
    if (n < NTAG) {
        float bo = bout[n];
        #pragma unroll
        for (int r = 0; r < 8; ++r) {
            int row = rowbase + r + (lane >> 4) * 8;
            feats[(size_t)row * NTAG + n] = acc[r] + bo;
        }
    }
}

// ---------------- CRF negative log-likelihood (one block per batch row) -----
__global__ void crf_nll_k(const float* __restrict__ feats,
                          const float* __restrict__ trans,
                          const int* __restrict__ blen,
                          const int* __restrict__ labels,
                          float* __restrict__ out_loss) {
    __shared__ float sT[NTAG * NTAG];
    __shared__ float alpha[NTAG];
    __shared__ float nalpha[NTAG];
    const int b = blockIdx.x, tid = threadIdx.x;
    for (int i = tid; i < NTAG * NTAG; i += blockDim.x) sT[i] = trans[i];
    __syncthreads();
    const float* fb = feats + (size_t)b * TT * NTAG;
    const int len = blen[b];
    if (tid < NTAG) alpha[tid] = fb[tid] + sT[TSTART * NTAG + tid];
    __syncthreads();
    for (int t = 1; t < TT; ++t) {
        if (t < len) {
            if (tid < NTAG) {
                float m = -3.0e38f;
                for (int i = 0; i < NTAG; ++i)
                    m = fmaxf(m, alpha[i] + sT[i * NTAG + tid]);
                float s = 0.0f;
                for (int i = 0; i < NTAG; ++i)
                    s += __expf(alpha[i] + sT[i * NTAG + tid] - m);
                nalpha[tid] = m + __logf(s) + fb[(size_t)t * NTAG + tid];
            }
            __syncthreads();
            if (tid < NTAG) alpha[tid] = nalpha[tid];
            __syncthreads();
        }
    }
    if (tid == 0) {
        float m = -3.0e38f;
        for (int j = 0; j < NTAG; ++j)
            m = fmaxf(m, alpha[j] + sT[j * NTAG + TSTOP]);
        float s = 0.0f;
        for (int j = 0; j < NTAG; ++j)
            s += __expf(alpha[j] + sT[j * NTAG + TSTOP] - m);
        float logZ = m + __logf(s);
        const int* lb = labels + (size_t)b * TT;
        float gold = sT[TSTART * NTAG + lb[0]];
        for (int t = 0; t < TT; ++t)
            if (t < len) gold += fb[(size_t)t * NTAG + lb[t]];
        for (int t = 1; t < TT; ++t)
            if (t < len) gold += sT[lb[t - 1] * NTAG + lb[t]];
        gold += sT[lb[len - 1] * NTAG + TSTOP];
        atomicAdd(out_loss, logZ - gold);
    }
}

// ---------------- Viterbi decode (one block per batch row) ------------------
__global__ void viterbi_k(const float* __restrict__ feats,
                          const float* __restrict__ trans,
                          const int* __restrict__ blen,
                          int* __restrict__ bp,
                          float* __restrict__ out) {   // out = d_out (tags at +1)
    __shared__ float sT[NTAG * NTAG];
    __shared__ float v[NTAG];
    __shared__ float nv[NTAG];
    const int b = blockIdx.x, tid = threadIdx.x;
    for (int i = tid; i < NTAG * NTAG; i += blockDim.x) sT[i] = trans[i];
    __syncthreads();
    const float* fb = feats + (size_t)b * TT * NTAG;
    int* bpb = bp + (size_t)b * TT * NTAG;
    const int len = blen[b];
    if (tid < NTAG) v[tid] = fb[tid] + sT[TSTART * NTAG + tid];
    __syncthreads();
    for (int t = 1; t < TT; ++t) {
        if (tid < NTAG) {
            if (t < len) {
                float m = -3.0e38f; int arg = 0;
                for (int i = 0; i < NTAG; ++i) {
                    float x = v[i] + sT[i * NTAG + tid];
                    if (x > m) { m = x; arg = i; }
                }
                nv[tid] = m + fb[(size_t)t * NTAG + tid];
                bpb[(size_t)t * NTAG + tid] = arg;
            } else {
                nv[tid] = v[tid];
                bpb[(size_t)t * NTAG + tid] = tid;
            }
        }
        __syncthreads();
        if (tid < NTAG) v[tid] = nv[tid];
        __syncthreads();
    }
    if (tid == 0) {
        float m = -3.0e38f; int best = 0;
        for (int j = 0; j < NTAG; ++j) {
            float x = v[j] + sT[j * NTAG + TSTOP];
            if (x > m) { m = x; best = j; }
        }
        float* ob = out + 1 + (size_t)b * TT;
        int cur = best;
        ob[TT - 1] = ((TT - 1) < len) ? (float)cur : 0.0f;
        for (int s = TT - 2; s >= 0; --s) {
            cur = bpb[(size_t)(s + 1) * NTAG + cur];
            ob[s] = (s < len) ? (float)cur : 0.0f;
        }
    }
}

// ---------------- host orchestration ---------------------------------------
extern "C" void kernel_launch(void* const* d_in, const int* in_sizes, int n_in,
                              void* d_out, int out_size, void* d_ws, size_t ws_size,
                              hipStream_t stream) {
    (void)in_sizes; (void)n_in; (void)out_size; (void)ws_size;
    const int*   batch_input   = (const int*)  d_in[0];
    const int*   batch_feature = (const int*)  d_in[1];
    const int*   batch_len     = (const int*)  d_in[2];
    /* d_in[3] batch_recover, d_in[4] mask: not needed (mask == t < len) */
    const int*   batch_label   = (const int*)  d_in[5];
    const float* char_emb      = (const float*)d_in[6];
    const float* feat_emb      = (const float*)d_in[7];
    const float* Wih_f         = (const float*)d_in[8];
    const float* Whh_f         = (const float*)d_in[9];
    const float* b_f           = (const float*)d_in[10];
    const float* Wih_b         = (const float*)d_in[11];
    const float* Whh_b         = (const float*)d_in[12];
    const float* b_b           = (const float*)d_in[13];
    const float* W_out         = (const float*)d_in[14];
    const float* b_out         = (const float*)d_in[15];
    const float* transitions   = (const float*)d_in[16];
    float* out = (float*)d_out;

    // workspace carve-up (256B aligned)
    char* p = (char*)d_ws;
    size_t off = 0;
    auto carve = [&](size_t bytes) {
        void* r = p + off;
        off += (bytes + 255) & ~size_t(255);
        return r;
    };
    unsigned short* xbf    = (unsigned short*)carve((size_t)BB * TT * DD * 2);
    unsigned short* wihf_b = (unsigned short*)carve((size_t)FOURH * DD * 2);
    unsigned short* whhf_b = (unsigned short*)carve((size_t)FOURH * HH * 2);
    unsigned short* wihb_b = (unsigned short*)carve((size_t)FOURH * DD * 2);
    unsigned short* whhb_b = (unsigned short*)carve((size_t)FOURH * HH * 2);
    unsigned short* woutb  = (unsigned short*)carve((size_t)80 * 1024 * 2);
    float*          hA     = (float*)carve((size_t)2 * BB * HH * 4);
    float*          hB     = (float*)carve((size_t)2 * BB * HH * 4);
    float*          cA     = (float*)carve((size_t)2 * BB * HH * 4);
    float*          cB     = (float*)carve((size_t)2 * BB * HH * 4);
    unsigned short* hbfA   = (unsigned short*)carve((size_t)2 * BB * HH * 2);
    unsigned short* hbfB   = (unsigned short*)carve((size_t)2 * BB * HH * 2);
    unsigned short* outf   = (unsigned short*)carve((size_t)BB * TT * HH * 2);
    unsigned short* outb   = (unsigned short*)carve((size_t)BB * TT * HH * 2);
    float*          feats  = (float*)carve((size_t)BB * TT * NTAG * 4);
    int*            bp     = (int*)carve((size_t)BB * TT * NTAG * 4);

    // 1) weight conversion to bf16
    f32_to_bf16_k<<<(FOURH * DD + 255) / 256, 256, 0, stream>>>(Wih_f, wihf_b, FOURH * DD);
    f32_to_bf16_k<<<(FOURH * HH + 255) / 256, 256, 0, stream>>>(Whh_f, whhf_b, FOURH * HH);
    f32_to_bf16_k<<<(FOURH * DD + 255) / 256, 256, 0, stream>>>(Wih_b, wihb_b, FOURH * DD);
    f32_to_bf16_k<<<(FOURH * HH + 255) / 256, 256, 0, stream>>>(Whh_b, whhb_b, FOURH * HH);
    conv_wout_k<<<(80 * 1024 + 255) / 256, 256, 0, stream>>>(W_out, woutb);

    // 2) embeddings
    {
        long long n = (long long)BB * TT * DD;
        embed_k<<<(unsigned)((n + 255) / 256), 256, 0, stream>>>(
            batch_input, batch_feature, char_emb, feat_emb, xbf);
    }

    // 3) init ping-pong h/c state + loss
    init_state_k<<<(2 * BB * HH) / 256, 256, 0, stream>>>(hA, hB, cA, cB, hbfA, hbfB, out);

    // 4) recurrent loop: 256 steps, both directions, single fused launch/step
    dim3 ggrid(HH / 32, BB / 16, 2);
    for (int t = 0; t < TT; ++t) {
        const bool even = ((t & 1) == 0);
        const float* h_in  = even ? hA : hB;
        const float* c_in  = even ? cA : cB;
        float*       h_out = even ? hB : hA;
        float*       c_out = even ? cB : cA;
        const unsigned short* hbf_in  = even ? hbfA : hbfB;
        unsigned short*       hbf_out = even ? hbfB : hbfA;
        lstm_step_fused_k<<<ggrid, 256, 0, stream>>>(
            xbf, hbf_in, wihf_b, whhf_b, wihb_b, whhb_b, b_f, b_b, batch_len,
            h_in, c_in, h_out, c_out, hbf_out, outf, outb, t);
    }

    // 5) emission projection
    emissions_gemm_k<<<dim3(512, 5), 256, 0, stream>>>(
        outf, outb, woutb, b_out, batch_len, feats);

    // 6) CRF NLL -> out[0]
    crf_nll_k<<<BB, 128, 0, stream>>>(feats, transitions, batch_len, batch_label, out);

    // 7) Viterbi -> out[1..B*T]
    viterbi_k<<<BB, 128, 0, stream>>>(feats, transitions, batch_len, bp, out);
}
